// Fusion_Block_20263655703077
// MI455X (gfx1250) — matrix operand order repaired and checked by reading, not compiled, not run on hardware
//
#include <hip/hip_runtime.h>
#include <math.h>

#define DIMX   64
#define HEADS  4
#define HD     16
#define HIDDEN 128
#define BB     8
#define NN     2048

typedef __attribute__((ext_vector_type(16))) _Float16 v16h;
typedef __attribute__((ext_vector_type(8)))  _Float16 v8h;
typedef __attribute__((ext_vector_type(8)))  float    v8f;

__device__ __forceinline__ v8f wmma16(v16h a, v16h b, v8f c) {
  return __builtin_amdgcn_wmma_f32_16x16x32_f16(false, a, false, b, (short)0, c, false, false);
}

// Load a 16x32 f16 A-fragment from a row-major LDS tile (stride in halves).
// Layout: lane L holds row (L&15); halves 0..15 -> K = kb..kb+15,
// kb = 16*(L>>4).
__device__ __forceinline__ v16h load_a_lds(const _Float16* tile, int stride, int lane, int ks) {
  int row = lane & 15;
  int kb  = (lane >> 4) * 16;
  const _Float16* p = tile + row * stride + ks * 32;
  v8h lo = *(const v8h*)(p + kb);
  v8h hi = *(const v8h*)(p + kb + 8);
  v16h a;
#pragma unroll
  for (int i = 0; i < 8; ++i) { a[i] = lo[i]; a[i + 8] = hi[i]; }
  return a;
}

// Pack a row-major fp32 weight [K][Ncols] into f16 B-fragments.
// B layout: lane L holds column (L&15), K rows 16*(L>>4)+h for halves h=0..15.
// Fragment order: frag = kt*(Ncols/16) + nt; per-frag 32 lanes * 16 halves.
__global__ void pack_b(const float* __restrict__ W, _Float16* __restrict__ out, int K, int Ncols) {
  int idx = blockIdx.x * 256 + threadIdx.x;
  int total = (K / 32) * (Ncols / 16) * 512;
  if (idx >= total) return;
  int h    = idx & 15;
  int lane = (idx >> 4) & 31;
  int frag = idx >> 9;
  int ntiles = Ncols / 16;
  int kt = frag / ntiles, nt = frag % ntiles;
  int col = nt * 16 + (lane & 15);
  int kk  = kt * 32 + (lane >> 4) * 16 + h;
  out[idx] = (_Float16)W[kk * Ncols + col];
}

// LayerNorm(x) @ Wkv -> K,Vt ; LayerNorm(clinic) @ Wq -> Q (scale folded).
__global__ void qkv_kernel(const float* __restrict__ x, const float* __restrict__ clinic,
                           const float* __restrict__ g1, const float* __restrict__ b1,
                           const float* __restrict__ bkv, const float* __restrict__ bq,
                           const _Float16* __restrict__ wq_p, const _Float16* __restrict__ wkv_p,
                           _Float16* __restrict__ Qd, _Float16* __restrict__ Kd,
                           _Float16* __restrict__ Vtd) {
  __shared__ _Float16 xn[16 * 64];
  __shared__ _Float16 cn[16 * 64];
  int lane = threadIdx.x;
  int tile = blockIdx.x;   // 0..127
  int b    = blockIdx.y;   // 0..7
  int row  = lane & 15;
  int half = lane >> 4;
  int tok  = tile * 16 + row;
  int c0   = half * 32;

  const float* srcs[2] = { x, clinic };
  _Float16*    dsts[2] = { xn, cn };
#pragma unroll
  for (int s = 0; s < 2; ++s) {
    const float* p = srcs[s] + ((size_t)b * NN + tok) * DIMX + c0;
    float v[32];
    float sum = 0.f, sq = 0.f;
#pragma unroll
    for (int i = 0; i < 32; ++i) { v[i] = p[i]; sum += v[i]; sq += v[i] * v[i]; }
    sum += __shfl_xor(sum, 16);
    sq  += __shfl_xor(sq, 16);
    float mu  = sum * (1.0f / 64.0f);
    float var = sq * (1.0f / 64.0f) - mu * mu;
    float rs  = rsqrtf(var + 1e-5f);
    _Float16* dst = dsts[s] + row * 64 + c0;
#pragma unroll
    for (int i = 0; i < 32; ++i)
      dst[i] = (_Float16)((v[i] - mu) * rs * g1[c0 + i] + b1[c0 + i]);
  }
  __syncthreads();

  const v16h* wq  = (const v16h*)wq_p;
  const v16h* wkv = (const v16h*)wkv_p;

  v16h acn0 = load_a_lds(cn, 64, lane, 0);
  v16h acn1 = load_a_lds(cn, 64, lane, 1);
  v16h axn0 = load_a_lds(xn, 64, lane, 0);
  v16h axn1 = load_a_lds(xn, 64, lane, 1);

  // ---- Q projection (rolled loop: one accumulator + one B-frag pair live) ----
#pragma unroll 1
  for (int nt = 0; nt < 4; ++nt) {
    v8f acc = {};
    acc = wmma16(acn0, wq[(0 * 4 + nt) * 32 + lane], acc);
    acc = wmma16(acn1, wq[(1 * 4 + nt) * 32 + lane], acc);
    int col = nt * 16 + row;
    float bias = bq[col];
    int head = nt, d = row;
    _Float16* qb = Qd + ((size_t)b * HEADS + head) * NN * HD;
#pragma unroll
    for (int r = 0; r < 8; ++r) {
      int t = tile * 16 + r + half * 8;
      qb[(size_t)t * HD + d] = (_Float16)((acc[r] + bias) * 0.25f); // fold 1/sqrt(hd)
    }
  }

  // ---- KV projection (rolled loop) ----
#pragma unroll 1
  for (int nt = 0; nt < 8; ++nt) {
    v8f acc = {};
    acc = wmma16(axn0, wkv[(0 * 8 + nt) * 32 + lane], acc);
    acc = wmma16(axn1, wkv[(1 * 8 + nt) * 32 + lane], acc);
    int col = nt * 16 + row;
    float bias = bkv[col];
    int head = nt & 3, d = row;
    if (nt < 4) {
      _Float16* kb = Kd + ((size_t)b * HEADS + head) * NN * HD;
#pragma unroll
      for (int r = 0; r < 8; ++r) {
        int t = tile * 16 + r + half * 8;
        kb[(size_t)t * HD + d] = (_Float16)(acc[r] + bias);
      }
    } else {
      _Float16* vb = Vtd + (((size_t)b * HEADS + head) * HD + d) * NN;
#pragma unroll
      for (int r = 0; r < 8; ++r) {
        int t = tile * 16 + r + half * 8;
        vb[t] = (_Float16)(acc[r] + bias);
      }
    }
  }
}

// Flash attention: 4 waves/block, each wave owns one 32-query tile
// (two 16-row A-fragments sharing every K/V fragment load).
__global__ void attn_kernel(const _Float16* __restrict__ Qd, const _Float16* __restrict__ Kd,
                            const _Float16* __restrict__ Vtd, float* __restrict__ o_ws) {
  __shared__ _Float16 plds[4][16][32];
  int lane = threadIdx.x & 31;
  int wid  = threadIdx.x >> 5;
  int h = blockIdx.y, b = blockIdx.z;
  int q0   = (blockIdx.x * 4 + wid) * 32;
  int col  = lane & 15;
  int half = lane >> 4;

  const _Float16* Qb = Qd + ((size_t)b * HEADS + h) * NN * HD;
  const _Float16* Kb = Kd + ((size_t)b * HEADS + h) * NN * HD;
  const _Float16* Vb = Vtd + ((size_t)b * HEADS + h) * HD * NN;

  // Two Q A-fragments: hd=16 padded to K=32 with zeros
  v16h aq[2];
  v8f  o[2];
  float m[2][8], l[2][8];
#pragma unroll
  for (int t = 0; t < 2; ++t) {
    v16h a = {};
    if (half == 0) a = *(const v16h*)(Qb + (size_t)(q0 + t * 16 + col) * HD);
    aq[t] = a;
    o[t] = (v8f){};
#pragma unroll
    for (int r = 0; r < 8; ++r) { m[t][r] = -INFINITY; l[t][r] = 0.f; }
  }

  for (int kt = 0; kt < NN / 32; ++kt) {
    int k0 = kt * 32;
    // B-fragments for K^T: only K-rows 0..15 (=hd) are real, rest zero.
    v16h bk0 = {}, bk1 = {};
    if (half == 0) {
      bk0 = *(const v16h*)(Kb + (size_t)(k0 + col) * HD);
      bk1 = *(const v16h*)(Kb + (size_t)(k0 + 16 + col) * HD);
    }
    v16h bv = *(const v16h*)(Vb + (size_t)col * NN + k0 + half * 16);

#pragma unroll
    for (int t = 0; t < 2; ++t) {
      v8f z0 = {}, z1 = {};
      v8f s0 = wmma16(aq[t], bk0, z0);
      v8f s1 = wmma16(aq[t], bk1, z1);

#pragma unroll
      for (int r = 0; r < 8; ++r) {
        float mx = fmaxf(s0[r], s1[r]);
#pragma unroll
        for (int ofs = 1; ofs < 16; ofs <<= 1) mx = fmaxf(mx, __shfl_xor(mx, ofs));
        float mnew = fmaxf(m[t][r], mx);
        float corr = __expf(m[t][r] - mnew);
        float p0 = __expf(s0[r] - mnew);
        float p1 = __expf(s1[r] - mnew);
        float rs = p0 + p1;
#pragma unroll
        for (int ofs = 1; ofs < 16; ofs <<= 1) rs += __shfl_xor(rs, ofs);
        l[t][r] = l[t][r] * corr + rs;
        o[t][r] *= corr;
        m[t][r] = mnew;
        plds[wid][r + half * 8][col]      = (_Float16)p0;
        plds[wid][r + half * 8][16 + col] = (_Float16)p1;
      }
      __syncthreads();
      v16h ap = load_a_lds(&plds[wid][0][0], 32, lane, 0);
      o[t] = wmma16(ap, bv, o[t]);
      __syncthreads();
    }
  }

  float* ob = o_ws + (size_t)b * NN * DIMX;
#pragma unroll
  for (int t = 0; t < 2; ++t) {
#pragma unroll
    for (int r = 0; r < 8; ++r) {
      int tk = q0 + t * 16 + r + half * 8;
      ob[(size_t)tk * DIMX + h * HD + col] = o[t][r] / l[t][r];
    }
  }
}

// out-proj + LayerNorm + MLP(GELU) + residual
__global__ void mlp_kernel(const float* __restrict__ o_ws, const float* __restrict__ bproj,
                           const float* __restrict__ g2, const float* __restrict__ b2n,
                           const float* __restrict__ b1v, const float* __restrict__ b2v,
                           const _Float16* __restrict__ wproj_p, const _Float16* __restrict__ w1_p,
                           const _Float16* __restrict__ w2_p, float* __restrict__ out) {
  __shared__ _Float16 lds[16 * 128];
  int lane = threadIdx.x;
  int tile = blockIdx.x;
  int b    = blockIdx.y;
  int col  = lane & 15;
  int half = lane >> 4;
  int tok  = tile * 16 + col;   // load row == lane&15
  int c0   = half * 32;

  // stage attention output tile as f16 A-source
  const float* p = o_ws + ((size_t)b * NN + tok) * DIMX + c0;
#pragma unroll
  for (int i = 0; i < 32; ++i) lds[col * 64 + c0 + i] = (_Float16)p[i];
  __syncthreads();

  const v16h* wp = (const v16h*)wproj_p;
  const v16h* w1 = (const v16h*)w1_p;
  const v16h* w2 = (const v16h*)w2_p;

  // out-projection (all four op tiles must stay live for the residual)
  float op[4][8];
  {
    v16h ao0 = load_a_lds(lds, 64, lane, 0);
    v16h ao1 = load_a_lds(lds, 64, lane, 1);
#pragma unroll
    for (int nt = 0; nt < 4; ++nt) {
      v8f acc = {};
      acc = wmma16(ao0, wp[(0 * 4 + nt) * 32 + lane], acc);
      acc = wmma16(ao1, wp[(1 * 4 + nt) * 32 + lane], acc);
      float bias = bproj[nt * 16 + col];
#pragma unroll
      for (int r = 0; r < 8; ++r) op[nt][r] = acc[r] + bias;
    }
  }

  // LayerNorm across 64 cols (4 fragments x 16 lanes per half)
  float mu[8], rstd[8];
#pragma unroll
  for (int r = 0; r < 8; ++r) {
    float s = op[0][r] + op[1][r] + op[2][r] + op[3][r];
    float q = op[0][r] * op[0][r] + op[1][r] * op[1][r] + op[2][r] * op[2][r] + op[3][r] * op[3][r];
#pragma unroll
    for (int ofs = 1; ofs < 16; ofs <<= 1) { s += __shfl_xor(s, ofs); q += __shfl_xor(q, ofs); }
    float m_ = s * (1.0f / 64.0f);
    float v_ = q * (1.0f / 64.0f) - m_ * m_;
    mu[r] = m_;
    rstd[r] = rsqrtf(v_ + 1e-5f);
  }
  __syncthreads();
#pragma unroll
  for (int nt = 0; nt < 4; ++nt) {
    int c = nt * 16 + col;
    float g = g2[c], bb = b2n[c];
#pragma unroll
    for (int r = 0; r < 8; ++r)
      lds[(r + half * 8) * 64 + c] = (_Float16)((op[nt][r] - mu[r]) * rstd[r] * g + bb);
  }
  __syncthreads();

  // MLP layer 1 + exact GELU (rolled loop to cap accumulator pressure)
  {
    v16h ah0 = load_a_lds(lds, 64, lane, 0);
    v16h ah1 = load_a_lds(lds, 64, lane, 1);
    __syncthreads();   // h1 overwrites the hn region below
#pragma unroll 1
    for (int nt = 0; nt < 8; ++nt) {
      v8f acc = {};
      acc = wmma16(ah0, w1[(0 * 8 + nt) * 32 + lane], acc);
      acc = wmma16(ah1, w1[(1 * 8 + nt) * 32 + lane], acc);
      int c = nt * 16 + col;
      float bias = b1v[c];
#pragma unroll
      for (int r = 0; r < 8; ++r) {
        float xv = acc[r] + bias;
        float gl = 0.5f * xv * (1.0f + erff(xv * 0.70710678118654752f));
        lds[(r + half * 8) * 128 + c] = (_Float16)gl;
      }
    }
  }
  __syncthreads();

  // MLP layer 2 + bias + residual (A-fragments re-read per nt, rolled loop)
#pragma unroll 1
  for (int nt = 0; nt < 4; ++nt) {
    v8f acc = {};
#pragma unroll
    for (int ks = 0; ks < 4; ++ks) {
      v16h a1 = load_a_lds(lds, 128, lane, ks);
      acc = wmma16(a1, w2[(ks * 4 + nt) * 32 + lane], acc);
    }
    int c = nt * 16 + col;
    float bias = b2v[c];
#pragma unroll
    for (int r = 0; r < 8; ++r) {
      int t = tile * 16 + r + half * 8;
      out[((size_t)b * NN + t) * DIMX + c] = acc[r] + bias + op[nt][r];
    }
  }
}

extern "C" void kernel_launch(void* const* d_in, const int* in_sizes, int n_in,
                              void* d_out, int out_size, void* d_ws, size_t ws_size,
                              hipStream_t stream) {
  const float* x      = (const float*)d_in[0];
  const float* clinic = (const float*)d_in[1];
  const float* n1g    = (const float*)d_in[2];
  const float* n1b    = (const float*)d_in[3];
  const float* n2g    = (const float*)d_in[4];
  const float* n2b    = (const float*)d_in[5];
  const float* Wkv    = (const float*)d_in[6];
  const float* bkv    = (const float*)d_in[7];
  const float* Wq     = (const float*)d_in[8];
  const float* bq     = (const float*)d_in[9];
  const float* Wproj  = (const float*)d_in[10];
  const float* bproj  = (const float*)d_in[11];
  const float* W1     = (const float*)d_in[12];
  const float* b1     = (const float*)d_in[13];
  const float* W2     = (const float*)d_in[14];
  const float* b2     = (const float*)d_in[15];

  char* ws = (char*)d_ws;
  _Float16* wq_p    = (_Float16*)(ws + 0);         // 8 frags  * 1KB = 8KB
  _Float16* wkv_p   = (_Float16*)(ws + 4096);      // 16 frags       = 16KB
  _Float16* wproj_p = (_Float16*)(ws + 12288);     // 8 frags        = 8KB
  _Float16* w1_p    = (_Float16*)(ws + 16384);     // 16 frags       = 16KB
  _Float16* w2_p    = (_Float16*)(ws + 24576);     // 16 frags       = 16KB
  const size_t qkvElems = (size_t)BB * HEADS * NN * HD;  // 1,048,576 halves = 2MB
  _Float16* Qd  = (_Float16*)(ws + 32768);
  _Float16* Kd  = Qd + qkvElems;
  _Float16* Vtd = Kd + qkvElems;
  float*    o_ws = (float*)(ws + 32768 + 3 * qkvElems * sizeof(_Float16));

  // Pack weights into WMMA B-fragment layout (f16)
  pack_b<<<16, 256, 0, stream>>>(Wq,    wq_p,    64, 64);
  pack_b<<<32, 256, 0, stream>>>(Wkv,   wkv_p,   64, 128);
  pack_b<<<16, 256, 0, stream>>>(Wproj, wproj_p, 64, 64);
  pack_b<<<32, 256, 0, stream>>>(W1,    w1_p,    64, 128);
  pack_b<<<32, 256, 0, stream>>>(W2,    w2_p,    128, 64);

  // LN + QKV projections (one wave per 16-token tile)
  qkv_kernel<<<dim3(NN / 16, BB), 32, 0, stream>>>(x, clinic, n1g, n1b, bkv, bq,
                                                   wq_p, wkv_p, Qd, Kd, Vtd);

  // Flash attention (4 waves/block, one 32-query tile per wave)
  attn_kernel<<<dim3(NN / 128, HEADS, BB), 128, 0, stream>>>(Qd, Kd, Vtd, o_ws);

  // out-proj + LN + MLP + residual
  mlp_kernel<<<dim3(NN / 16, BB), 32, 0, stream>>>(o_ws, bproj, n2g, n2b, b1, b2,
                                                   wproj_p, w1_p, w2_p, (float*)d_out);
}
